// LagCrossNodeAttention_71760313582000
// MI455X (gfx1250) — matrix operand order repaired and checked by reading, not compiled, not run on hardware
//
#include <hip/hip_runtime.h>
#include <cstdint>

// ---------------------------------------------------------------------------
// LagCrossNodeAttention for MI455X (gfx1250, wave32, WMMA).
// All GEMM stages use v_wmma_f32_16x16x32_bf16 (fp32 accumulate).
// Shared A-tiles are staged to LDS with gfx1250 async global->LDS DMA.
// ---------------------------------------------------------------------------

typedef __attribute__((ext_vector_type(16))) __bf16    v16bf;
typedef __attribute__((ext_vector_type(8)))  float     v8f;
typedef __attribute__((ext_vector_type(4)))  unsigned  uivec4;

#define B_   2
#define T_   128
#define N_   128
#define KL_  3
#define NH_  4
#define DH_  32
#define D_   128
#define TR_  (T_ - KL_)               // 125
#define ROWS_ALL (B_*T_*N_)           // 32768
#define ROWS_TR  (B_*TR_*N_)          // 32000
#define EPS_ 1e-8f

union U16bf { v16bf v; uivec4 q[2]; };

// ---- gfx1250 async global->LDS copy (16B per lane), ASYNCcnt tracked ------
__device__ __forceinline__ void async_copy_b128(void* lds_dst, const void* gsrc) {
  // Low 32 bits of a generic LDS pointer == LDS byte address (ISA aperture rule).
  unsigned ldsa = (unsigned)(uintptr_t)lds_dst;
  asm volatile("global_load_async_to_lds_b128 %0, %1, off"
               :: "v"(ldsa), "v"(gsrc)
               : "memory");
}
__device__ __forceinline__ void async_wait0() {
  asm volatile("s_wait_asynccnt 0x0" ::: "memory");
}

// A-matrix (16x32 bf16) fragment: lane L holds row M=L&15;
// K range = (L<16 ? 0..15 : 16..31), 16 contiguous halves along K.
__device__ __forceinline__ v16bf load_afrag(const __bf16* p /* row start + k-origin */) {
  const int lane = threadIdx.x & 31;
  const int kb = (lane & 16) ? 16 : 0;
  U16bf u;
  u.q[0] = *reinterpret_cast<const uivec4*>(p + kb);
  u.q[1] = *reinterpret_cast<const uivec4*>(p + kb + 8);
  return u.v;
}
// B-matrix (32x16 bf16) fragment: lane L holds col N=L&15;
// K range = (L<16 ? 0..15 : 16..31), 16 contiguous halves along K.
__device__ __forceinline__ v16bf load_bfrag(const __bf16* p /* row(n) start + k-origin */) {
  const int lane = threadIdx.x & 31;
  const int kb = (lane & 16) ? 16 : 0;
  U16bf u;
  u.q[0] = *reinterpret_cast<const uivec4*>(p + kb);
  u.q[1] = *reinterpret_cast<const uivec4*>(p + kb + 8);
  return u.v;
}
__device__ __forceinline__ v8f wmma_bf16(v16bf a, v16bf b, v8f c) {
  return __builtin_amdgcn_wmma_f32_16x16x32_bf16(false, a, false, b, (short)0, c, false, false);
}
// reductions inside a 16-lane half (C-layout rows live per half)
__device__ __forceinline__ float half16_sum(float v) {
#pragma unroll
  for (int m = 1; m < 16; m <<= 1) v += __shfl_xor(v, m, 32);
  return v;
}
__device__ __forceinline__ float half16_max(float v) {
#pragma unroll
  for (int m = 1; m < 16; m <<= 1) v = fmaxf(v, __shfl_xor(v, m, 32));
  return v;
}
__device__ __forceinline__ float wave_sum(float v) {
#pragma unroll
  for (int m = 1; m < 32; m <<= 1) v += __shfl_xor(v, m, 32);
  return v;
}

// ------------------------------- init kernels ------------------------------
__global__ void k_cvt_bf16(const float* __restrict__ s, __bf16* __restrict__ d, int n) {
  int i = blockIdx.x * blockDim.x + threadIdx.x;
  if (i < n) d[i] = (__bf16)s[i];
}
__global__ void k_transpose_bf16(const float* __restrict__ W, __bf16* __restrict__ WT,
                                 int K, int Ncol) {
  int i = blockIdx.x * blockDim.x + threadIdx.x;
  if (i < K * Ncol) { int k = i / Ncol, n = i % Ncol; WT[n * K + k] = (__bf16)W[k * Ncol + n]; }
}
__global__ void k_logA(const float* __restrict__ A, float* __restrict__ logA,
                       float* __restrict__ rsA, float* __restrict__ logrsA) {
  int row = blockIdx.x;                 // (b,k,i) in [0, B*KL*N)
  int j = threadIdx.x;
  float a = A[row * N_ + j];
  logA[row * N_ + j] = logf(a + EPS_);
  __shared__ float red[128];
  red[j] = a; __syncthreads();
  for (int s = 64; s > 0; s >>= 1) { if (j < s) red[j] += red[j + s]; __syncthreads(); }
  if (j == 0) { rsA[row] = red[0]; logrsA[row] = logf(red[0] + EPS_); }
}
__global__ void k_zero(float* p, int n) {
  int i = blockIdx.x * blockDim.x + threadIdx.x; if (i < n) p[i] = 0.f;
}

// ------------------------- QKV projection (WMMA) ---------------------------
// A-tile (16 rows x 128) is shared by all 8 waves: DMA it into LDS once.
__global__ void __launch_bounds__(256)
k_qkv(const __bf16* __restrict__ Hbf,
      const __bf16* __restrict__ WqT, const __bf16* __restrict__ WkT,
      const __bf16* __restrict__ WvT,
      __bf16* __restrict__ Qbf, __bf16* __restrict__ Kbf, __bf16* __restrict__ VbfT) {
  __shared__ __bf16 sA[16 * D_];        // 4 KB
  const int wave = threadIdx.x >> 5, lane = threadIdx.x & 31;
  const int lane15 = lane & 15, hi8 = (lane & 16) ? 8 : 0;
  const int r0 = blockIdx.x * 16;       // row tile (32768 rows -> 2048 tiles)
  const int n0 = wave * 16;             // 8 waves cover 128 cols
  const int sel = blockIdx.y;
  const __bf16* WT = sel == 0 ? WqT : (sel == 1 ? WkT : WvT);
  { // 256 x 16B async chunks (one per thread)
    int ch = threadIdx.x;               // row = ch>>4, col chunk = ch&15
    int row = ch >> 4, cc = (ch & 15) * 8;
    async_copy_b128(sA + row * D_ + cc, Hbf + (size_t)(r0 + row) * D_ + cc);
    async_wait0();
  }
  __syncthreads();
  v8f c = {0.f,0.f,0.f,0.f,0.f,0.f,0.f,0.f};
#pragma unroll
  for (int kc = 0; kc < 4; kc++) {
    v16bf a = load_afrag(sA + lane15 * D_ + kc * 32);
    v16bf b = load_bfrag(WT + (size_t)(n0 + lane15) * D_ + kc * 32);
    c = wmma_bf16(a, b, c);
  }
  const int col = n0 + lane15;
#pragma unroll
  for (int r = 0; r < 8; r++) {
    int row = r0 + r + hi8;
    __bf16 v = (__bf16)c[r];
    if (sel == 0)      Qbf[(size_t)row * D_ + col] = v;
    else if (sel == 1) Kbf[(size_t)row * D_ + col] = v;
    else {
      int bt = row >> 7, i = row & 127;        // N_=128
      int h = col >> 5, d = col & 31;          // DH_=32
      VbfT[(((size_t)bt * NH_ + h) * DH_ + d) * N_ + i] = v;  // transposed store
    }
  }
}

// ------------------------ means / context projections ----------------------
__global__ void k_hmean(const float* __restrict__ H, float* __restrict__ Hm) {
  int bt = blockIdx.x, d = threadIdx.x;
  float s = 0.f;
  for (int i = 0; i < N_; i++) s += H[((size_t)bt * N_ + i) * D_ + d];
  Hm[bt * D_ + d] = s * (1.f / N_);
}
__global__ void k_cqk(const float* __restrict__ Hm, const float* __restrict__ Wq,
                      const float* __restrict__ Wk, float* __restrict__ CQ,
                      float* __restrict__ CK) {
  int bt = blockIdx.x, c = threadIdx.x;
  float sq = 0.f, sk = 0.f;
  for (int k = 0; k < D_; k++) {
    float hv = Hm[bt * D_ + k];
    sq += hv * Wq[k * D_ + c];
    sk += hv * Wk[k * D_ + c];
  }
  CQ[bt * D_ + c] = sq; CK[bt * D_ + c] = sk;
}
__global__ void k_projq(const __bf16* __restrict__ Qbf, const float* __restrict__ CQ,
                        const float* __restrict__ rho, float* __restrict__ projq,
                        float* __restrict__ qrho) {
  int t = blockIdx.x * blockDim.x + threadIdx.x;   // (row,h)
  if (t >= ROWS_ALL * NH_) return;
  int row = t >> 2, h = t & 3;
  int bt = row >> 7;
  float q[DH_]; float pq = 0.f;
#pragma unroll
  for (int d = 0; d < DH_; d++) {
    q[d] = (float)Qbf[(size_t)row * D_ + h * DH_ + d];
    pq += q[d] * CQ[bt * D_ + h * DH_ + d];
  }
  projq[t] = pq;
#pragma unroll
  for (int k = 0; k < KL_; k++) {
    float s = 0.f;
#pragma unroll
    for (int d = 0; d < DH_; d++) s += q[d] * rho[k * DH_ + d];
    qrho[(size_t)k * ROWS_ALL * NH_ + t] = s;   // Q . rho[k]  (folds rho out of K loads)
  }
}
__global__ void k_projk(const __bf16* __restrict__ Kbf, const float* __restrict__ CK,
                        const float* __restrict__ rho, float* __restrict__ projk) {
  long t = (long)blockIdx.x * blockDim.x + threadIdx.x;  // (k,row,h)
  if (t >= (long)KL_ * ROWS_ALL * NH_) return;
  int k = (int)(t / (ROWS_ALL * NH_));
  int rh = (int)(t % (ROWS_ALL * NH_));
  int row = rh >> 2, h = rh & 3;
  int bt = row >> 7;
  float s = 0.f;
#pragma unroll
  for (int d = 0; d < DH_; d++) {
    float kv = (float)Kbf[(size_t)row * D_ + h * DH_ + d] + rho[k * DH_ + d];
    float cv = CK[bt * D_ + h * DH_ + d] + rho[k * DH_ + d];
    s += kv * cv;
  }
  projk[t] = s;
}

// -------------------- fused lag attention (WMMA + softmax) -----------------
// One block per (lag k, b, t); 8 waves own 16-row strips; loop over 4 heads.
__global__ void __launch_bounds__(256)
k_attn(const __bf16* __restrict__ Qbf, const __bf16* __restrict__ Kbf,
       const __bf16* __restrict__ VbfT, const float* __restrict__ projq,
       const float* __restrict__ projk, const float* __restrict__ qrho,
       const float* __restrict__ logA, const float* __restrict__ A,
       const float* __restrict__ rsA, const float* __restrict__ logrsA,
       const float* __restrict__ beta, const float* __restrict__ gamma_p,
       const float* __restrict__ lambda_p, float* __restrict__ lag,
       __bf16* __restrict__ msg, float* __restrict__ lossS) {
  __shared__ __bf16 sp[N_ * N_];   // spatial = mean_h(omega)  (32 KB)
  __shared__ __bf16 om[N_ * N_];   // omega staging for msg GEMM (32 KB)
  const int wave = threadIdx.x >> 5, lane = threadIdx.x & 31;
  const int lane15 = lane & 15, hi8 = (lane & 16) ? 8 : 0;
  const int blk = blockIdx.x;                    // kk*(B*TR) + b*TR + tt
  const int kk = blk / (B_ * TR_);
  const int rem = blk % (B_ * TR_);
  const int b = rem / TR_, tt = rem % TR_;
  const int tq = KL_ + tt, s_t = tq - (kk + 1);
  const int qrow0 = (b * T_ + tq) * N_;
  const int srow0 = (b * T_ + s_t) * N_;
  const int i0 = wave * 16;
  const float scale = rsqrtf((float)DH_);
  const float betak = beta[kk];
  const float gam = gamma_p[0], lam = lambda_p[0];

  { // zero this wave's spatial rows
    unsigned* spw = reinterpret_cast<unsigned*>(sp + i0 * N_);
    for (int x = lane; x < 16 * N_ / 2; x += 32) spw[x] = 0u;
  }
  __syncthreads();

  float lagacc[8];
#pragma unroll
  for (int r = 0; r < 8; r++) lagacc[r] = 0.f;

  for (int h = 0; h < NH_; h++) {
    v16bf qa = load_afrag(Qbf + (size_t)(qrow0 + i0 + lane15) * D_ + h * DH_);
    float pqv[8], qrv[8];
#pragma unroll
    for (int r = 0; r < 8; r++) {
      int irow = qrow0 + i0 + r + hi8;
      pqv[r] = projq[(size_t)irow * NH_ + h];
      qrv[r] = qrho[(size_t)kk * ROWS_ALL * NH_ + (size_t)irow * NH_ + h];
    }
    float lt[8][8];                    // logits: 8 col-tiles x 8 C rows
#pragma unroll
    for (int jt = 0; jt < 8; jt++) {
      v16bf kb = load_bfrag(Kbf + (size_t)(srow0 + jt * 16 + lane15) * D_ + h * DH_);
      v8f c = {0.f,0.f,0.f,0.f,0.f,0.f,0.f,0.f};
      c = wmma_bf16(qa, kb, c);        // K = dh = 32: exactly one WMMA
      int jcol = jt * 16 + lane15;
      float pkj = projk[((size_t)kk * ROWS_ALL + srow0 + jcol) * NH_ + h];
#pragma unroll
      for (int r = 0; r < 8; r++) {
        int i = i0 + r + hi8;
        float la = logA[((size_t)(b * KL_ + kk) * N_ + i) * N_ + jcol];
        float l = scale * (c[r] + qrv[r]) + betak
                - lam * scale * pqv[r] * pkj + gam * la;
        lt[jt][r] = l;
        lagacc[r] += l;
      }
    }
    // row softmax (row spread over 8 tiles x 16 lanes of this half)
#pragma unroll
    for (int r = 0; r < 8; r++) {
      float m = lt[0][r];
#pragma unroll
      for (int jt = 1; jt < 8; jt++) m = fmaxf(m, lt[jt][r]);
      m = half16_max(m);
      float s = 0.f;
#pragma unroll
      for (int jt = 0; jt < 8; jt++) { float e = __expf(lt[jt][r] - m); lt[jt][r] = e; s += e; }
      s = half16_sum(s);
      float inv = 1.f / s;
      int i = i0 + r + hi8;
#pragma unroll
      for (int jt = 0; jt < 8; jt++) {
        float w = lt[jt][r] * inv;
        int idx = i * N_ + jt * 16 + lane15;
        om[idx] = (__bf16)w;
        sp[idx] = (__bf16)((float)sp[idx] + 0.25f * w);   // mean over 4 heads
      }
    }
    __syncthreads();   // make omega visible / order LDS traffic vs fragment reads
    // msg = omega @ V   (16 x 32 per wave; K = 128 -> 4 WMMAs per 16-col tile)
#pragma unroll
    for (int nt = 0; nt < 2; nt++) {
      v8f mc = {0.f,0.f,0.f,0.f,0.f,0.f,0.f,0.f};
#pragma unroll
      for (int kc = 0; kc < 4; kc++) {
        v16bf ma = load_afrag(om + (i0 + lane15) * N_ + kc * 32);
        v16bf mb = load_bfrag(VbfT + (((size_t)(b * T_ + s_t) * NH_ + h) * DH_
                                      + nt * 16 + lane15) * N_ + kc * 32);
        mc = wmma_bf16(ma, mb, mc);
      }
#pragma unroll
      for (int r = 0; r < 8; r++) {
        int i = i0 + r + hi8;
        msg[((size_t)blk * N_ + i) * D_ + h * DH_ + nt * 16 + lane15] = (__bf16)mc[r];
      }
    }
    __syncthreads();
  }
  // lag_score = mean over (h, j)
#pragma unroll
  for (int r = 0; r < 8; r++) {
    float v = half16_sum(lagacc[r]);
    if (lane15 == 0) {
      int i = i0 + r + hi8;
      lag[(size_t)blk * N_ + i] = v * (1.f / (NH_ * N_));
    }
  }
  // loss_s contribution: sum_ij A_sp * (log A_sp - log(spatial + eps))
  float lacc = 0.f;
  for (int rr = 0; rr < 16; rr++) {
    int i = i0 + rr;
    int arow = (b * KL_ + kk) * N_ + i;
    float inv_rs = 1.f / (rsA[arow] + EPS_);
    float lrs = logrsA[arow];
    for (int j = lane; j < N_; j += 32) {
      float a = A[(size_t)arow * N_ + j];
      float spv = (float)sp[i * N_ + j];
      lacc += (a * inv_rs) * (logA[(size_t)arow * N_ + j] - lrs - logf(spv + EPS_));
    }
  }
  lacc = wave_sum(lacc);
  if (lane == 0) atomicAdd(lossS, lacc);
}

// ------------------- pi softmax + lag combine + loss_t ---------------------
__global__ void k_combine(const float* __restrict__ lag, const __bf16* __restrict__ msg,
                          const float* __restrict__ rsA, __bf16* __restrict__ msgc,
                          float* __restrict__ lossT) {
  int t = blockIdx.x * blockDim.x + threadIdx.x;
  if (t >= ROWS_TR * D_) return;
  int row = t >> 7, c = t & 127;
  int b = row / (TR_ * N_);
  int ttn = row % (TR_ * N_);                 // tt*N + i
  float lv[KL_];
#pragma unroll
  for (int k = 0; k < KL_; k++)
    lv[k] = lag[(size_t)k * (B_ * TR_ * N_) + (size_t)b * TR_ * N_ + ttn];
  float m = fmaxf(lv[0], fmaxf(lv[1], lv[2]));
  float e[KL_]; float s = 0.f;
#pragma unroll
  for (int k = 0; k < KL_; k++) { e[k] = __expf(lv[k] - m); s += e[k]; }
  float inv = 1.f / s;
  float acc = 0.f;
#pragma unroll
  for (int k = 0; k < KL_; k++)
    acc += (e[k] * inv) *
           (float)msg[((size_t)k * (B_ * TR_ * N_) + (size_t)b * TR_ * N_ + ttn) * D_ + c];
  msgc[(size_t)row * D_ + c] = (__bf16)acc;
  if (c == 0) {
    int i = ttn % N_;
    float rk[KL_]; float S = 0.f;
#pragma unroll
    for (int k = 0; k < KL_; k++) { rk[k] = rsA[(b * KL_ + k) * N_ + i]; S += rk[k]; }
    float invS = 1.f / (S + EPS_);
    float ltv = 0.f;
#pragma unroll
    for (int k = 0; k < KL_; k++) {
      float tgt = rk[k] * invS;
      ltv += tgt * (logf(tgt) - logf(e[k] * inv + EPS_));
    }
    atomicAdd(lossT, ltv);
  }
}

// ---------------------------- msg_all @ Wo (WMMA) --------------------------
__global__ void __launch_bounds__(256)
k_gemm_mo(const __bf16* __restrict__ Am, const __bf16* __restrict__ WoT,
          float* __restrict__ mo) {
  __shared__ __bf16 sA[16 * D_];        // 4 KB, shared A-tile
  const int wave = threadIdx.x >> 5, lane = threadIdx.x & 31;
  const int lane15 = lane & 15, hi8 = (lane & 16) ? 8 : 0;
  const int r0 = blockIdx.x * 16, n0 = wave * 16;
  { // async DMA A-tile to LDS
    int ch = threadIdx.x;
    int row = ch >> 4, cc = (ch & 15) * 8;
    async_copy_b128(sA + row * D_ + cc, Am + (size_t)(r0 + row) * D_ + cc);
    async_wait0();
  }
  __syncthreads();
  v8f c = {0.f,0.f,0.f,0.f,0.f,0.f,0.f,0.f};
#pragma unroll
  for (int kc = 0; kc < 4; kc++) {
    v16bf a = load_afrag(sA + lane15 * D_ + kc * 32);
    v16bf b = load_bfrag(WoT + (size_t)(n0 + lane15) * D_ + kc * 32);
    c = wmma_bf16(a, b, c);
  }
#pragma unroll
  for (int r = 0; r < 8; r++)
    mo[(size_t)(r0 + r + hi8) * D_ + n0 + lane15] = c[r];
}

// --------------------------------- gate ------------------------------------
__global__ void k_gate(const float* __restrict__ H, const float* __restrict__ mo,
                       const float* __restrict__ gw, const float* __restrict__ gb,
                       __bf16* __restrict__ gm) {
  int row = blockIdx.x * blockDim.x + threadIdx.x;
  if (row >= ROWS_TR) return;
  int b = row / (TR_ * N_);
  int rr = row % (TR_ * N_);
  int tt = rr / N_, i = rr % N_;
  size_t hrow = ((size_t)(b * T_ + KL_ + tt) * N_ + i) * D_;
  float z = gb[0];
  for (int c = 0; c < D_; c++) z += H[hrow + c] * gw[c];
  for (int c = 0; c < D_; c++) z += mo[(size_t)row * D_ + c] * gw[D_ + c];
  z = 1.f / (1.f + __expf(-z));
  for (int c = 0; c < D_; c++)
    gm[(size_t)row * D_ + c] = (__bf16)(z * mo[(size_t)row * D_ + c]);
}

// ------------------------------- FFN1 (WMMA) -------------------------------
// A = concat([H_base, z*msg_all]); staged from two bf16 sources into LDS
// via async DMA, then consumed as fragments; K = 256.
__global__ void __launch_bounds__(256)
k_ffn1(const __bf16* __restrict__ Hbf, const __bf16* __restrict__ gm,
       const __bf16* __restrict__ W1T, const float* __restrict__ b1,
       __bf16* __restrict__ ff1) {
  __shared__ __bf16 sA[16 * 256];       // 8 KB concat A-tile
  const int wave = threadIdx.x >> 5, lane = threadIdx.x & 31;
  const int lane15 = lane & 15, hi8 = (lane & 16) ? 8 : 0;
  const int r0 = blockIdx.x * 16;
  const int n0 = blockIdx.y * 128 + wave * 16;
  { // 512 x 16B chunks: cc<16 from H_base (bf16), cc>=16 from z*msg_all
    for (int ch = threadIdx.x; ch < 512; ch += 256) {
      int row = ch >> 5, cc = ch & 31;
      int rowA = r0 + row;
      int b = rowA / (TR_ * N_);
      int rr = rowA % (TR_ * N_);
      int tt = rr / N_, i = rr % N_;
      const __bf16* src = (cc < 16)
          ? Hbf + ((size_t)(b * T_ + KL_ + tt) * N_ + i) * D_ + cc * 8
          : gm + (size_t)rowA * D_ + (cc - 16) * 8;
      async_copy_b128(sA + row * 256 + cc * 8, src);
    }
    async_wait0();
  }
  __syncthreads();
  v8f c = {0.f,0.f,0.f,0.f,0.f,0.f,0.f,0.f};
#pragma unroll
  for (int kc = 0; kc < 8; kc++) {
    v16bf a = load_afrag(sA + lane15 * 256 + kc * 32);
    v16bf bf = load_bfrag(W1T + (size_t)(n0 + lane15) * 256 + kc * 32);
    c = wmma_bf16(a, bf, c);
  }
  int col = n0 + lane15;
  float bias = b1[col];
#pragma unroll
  for (int r = 0; r < 8; r++)
    ff1[(size_t)(r0 + r + hi8) * 256 + col] = (__bf16)fmaxf(c[r] + bias, 0.f);
}

// -------------------- FFN2 + residual + LayerNorm (WMMA) -------------------
// Each wave owns 16 full rows (all 8 col tiles) so LN row stats stay in-wave.
__global__ void __launch_bounds__(256)
k_ffn2_ln(const __bf16* __restrict__ ff1, const __bf16* __restrict__ W2T,
          const float* __restrict__ b2, const float* __restrict__ H,
          const float* __restrict__ ln_g, const float* __restrict__ ln_b,
          float* __restrict__ out) {
  const int wave = threadIdx.x >> 5, lane = threadIdx.x & 31;
  const int lane15 = lane & 15, hi8 = (lane & 16) ? 8 : 0;
  const int r0 = blockIdx.x * 128 + wave * 16;
  float x[8][8];
#pragma unroll
  for (int nt = 0; nt < 8; nt++) {
    v8f c = {0.f,0.f,0.f,0.f,0.f,0.f,0.f,0.f};
#pragma unroll
    for (int kc = 0; kc < 8; kc++) {
      v16bf a = load_afrag(ff1 + (size_t)(r0 + lane15) * 256 + kc * 32);
      v16bf b = load_bfrag(W2T + (size_t)(nt * 16 + lane15) * 256 + kc * 32);
      c = wmma_bf16(a, b, c);
    }
#pragma unroll
    for (int r = 0; r < 8; r++) x[nt][r] = c[r];
  }
#pragma unroll
  for (int r = 0; r < 8; r++) {
    int row = r0 + r + hi8;
    int b = row / (TR_ * N_);
    int rr = row % (TR_ * N_);
    int tt = rr / N_, i = rr % N_;
    size_t orow = ((size_t)(b * T_ + KL_ + tt) * N_ + i) * D_;
    float s1 = 0.f, s2 = 0.f, xv[8];
#pragma unroll
    for (int nt = 0; nt < 8; nt++) {
      int col = nt * 16 + lane15;
      float v = x[nt][r] + b2[col] + H[orow + col];
      xv[nt] = v; s1 += v; s2 += v * v;
    }
    s1 = half16_sum(s1); s2 = half16_sum(s2);
    float mean = s1 * (1.f / D_);
    float var = s2 * (1.f / D_) - mean * mean;
    float rstd = rsqrtf(var + 1e-5f);
#pragma unroll
    for (int nt = 0; nt < 8; nt++) {
      int col = nt * 16 + lane15;
      out[orow + col] = ln_g[col] * (xv[nt] - mean) * rstd + ln_b[col];
    }
  }
}

// ------------------------ head copy + loss finalize ------------------------
__global__ void k_head_loss(const float* __restrict__ H, const float* __restrict__ lossT,
                            const float* __restrict__ lossS, float* __restrict__ out) {
  int t = blockIdx.x * blockDim.x + threadIdx.x;
  const int nh = B_ * KL_ * N_ * D_;
  if (t < nh) {
    int b = t / (KL_ * N_ * D_);
    int rem = t % (KL_ * N_ * D_);
    size_t idx = (size_t)b * T_ * N_ * D_ + rem;   // D_IN == D_OUT: head = H[:, :K]
    out[idx] = H[idx];
  }
  if (t == 0)
    out[(size_t)B_ * T_ * N_ * D_] =
        0.5f * (lossT[0] / B_) + 0.5f * (lossS[0] / B_);
}

// =============================== launcher ==================================
extern "C" void kernel_launch(void* const* d_in, const int* in_sizes, int n_in,
                              void* d_out, int out_size, void* d_ws, size_t ws_size,
                              hipStream_t stream) {
  (void)in_sizes; (void)n_in; (void)out_size; (void)ws_size;
  const float* H       = (const float*)d_in[0];
  const float* A       = (const float*)d_in[1];
  const float* Wq      = (const float*)d_in[2];
  const float* Wk      = (const float*)d_in[3];
  const float* Wv      = (const float*)d_in[4];
  const float* Wo      = (const float*)d_in[5];
  const float* rho     = (const float*)d_in[6];
  const float* beta    = (const float*)d_in[7];
  const float* gate_w  = (const float*)d_in[8];
  const float* gate_b  = (const float*)d_in[9];
  const float* ffn_w1  = (const float*)d_in[10];
  const float* ffn_b1  = (const float*)d_in[11];
  const float* ffn_w2  = (const float*)d_in[12];
  const float* ffn_b2  = (const float*)d_in[13];
  const float* ln_g    = (const float*)d_in[14];
  const float* ln_b    = (const float*)d_in[15];
  const float* gamma_A = (const float*)d_in[16];
  const float* lambda_b= (const float*)d_in[17];
  float* out = (float*)d_out;

  char* base = (char*)d_ws; size_t off = 0;
  auto alloc = [&](size_t bytes) -> void* {
    off = (off + 255) & ~(size_t)255;
    void* p = base + off; off += bytes; return p;
  };
  __bf16* Hbf   = (__bf16*)alloc((size_t)ROWS_ALL * D_ * 2);
  __bf16* Qbf   = (__bf16*)alloc((size_t)ROWS_ALL * D_ * 2);
  __bf16* Kbf   = (__bf16*)alloc((size_t)ROWS_ALL * D_ * 2);
  __bf16* VbfT  = (__bf16*)alloc((size_t)ROWS_ALL * D_ * 2);
  __bf16* WqT   = (__bf16*)alloc(D_ * D_ * 2);
  __bf16* WkT   = (__bf16*)alloc(D_ * D_ * 2);
  __bf16* WvT   = (__bf16*)alloc(D_ * D_ * 2);
  __bf16* WoT   = (__bf16*)alloc(D_ * D_ * 2);
  __bf16* W1T   = (__bf16*)alloc(256 * 256 * 2);
  __bf16* W2T   = (__bf16*)alloc(128 * 256 * 2);
  float*  Hm    = (float*)alloc((size_t)B_ * T_ * D_ * 4);
  float*  CQ    = (float*)alloc((size_t)B_ * T_ * D_ * 4);
  float*  CK    = (float*)alloc((size_t)B_ * T_ * D_ * 4);
  float*  projq = (float*)alloc((size_t)ROWS_ALL * NH_ * 4);
  float*  projk = (float*)alloc((size_t)KL_ * ROWS_ALL * NH_ * 4);
  float*  qrho  = (float*)alloc((size_t)KL_ * ROWS_ALL * NH_ * 4);
  float*  logA  = (float*)alloc((size_t)B_ * KL_ * N_ * N_ * 4);
  float*  rsA   = (float*)alloc((size_t)B_ * KL_ * N_ * 4);
  float*  logrsA= (float*)alloc((size_t)B_ * KL_ * N_ * 4);
  float*  lag   = (float*)alloc((size_t)KL_ * B_ * TR_ * N_ * 4);
  __bf16* msg   = (__bf16*)alloc((size_t)KL_ * ROWS_TR * D_ * 2);
  __bf16* msgc  = (__bf16*)alloc((size_t)ROWS_TR * D_ * 2);
  float*  mo    = (float*)alloc((size_t)ROWS_TR * D_ * 4);
  __bf16* gm    = (__bf16*)alloc((size_t)ROWS_TR * D_ * 2);
  __bf16* ff1   = (__bf16*)alloc((size_t)ROWS_TR * 256 * 2);
  float*  loss  = (float*)alloc(2 * 4);    // [lossT, lossS]

  // ---- init ----
  k_cvt_bf16<<<(ROWS_ALL * D_ + 255) / 256, 256, 0, stream>>>(H, Hbf, ROWS_ALL * D_);
  k_transpose_bf16<<<(D_ * D_ + 255) / 256, 256, 0, stream>>>(Wq, WqT, D_, D_);
  k_transpose_bf16<<<(D_ * D_ + 255) / 256, 256, 0, stream>>>(Wk, WkT, D_, D_);
  k_transpose_bf16<<<(D_ * D_ + 255) / 256, 256, 0, stream>>>(Wv, WvT, D_, D_);
  k_transpose_bf16<<<(D_ * D_ + 255) / 256, 256, 0, stream>>>(Wo, WoT, D_, D_);
  k_transpose_bf16<<<(256 * 256 + 255) / 256, 256, 0, stream>>>(ffn_w1, W1T, 256, 256);
  k_transpose_bf16<<<(256 * 128 + 255) / 256, 256, 0, stream>>>(ffn_w2, W2T, 256, 128);
  k_logA<<<B_ * KL_ * N_, 128, 0, stream>>>(A, logA, rsA, logrsA);
  k_zero<<<1, 32, 0, stream>>>(loss, 2);
  // ---- projections ----
  k_qkv<<<dim3(ROWS_ALL / 16, 3), 256, 0, stream>>>(Hbf, WqT, WkT, WvT, Qbf, Kbf, VbfT);
  k_hmean<<<B_ * T_, 128, 0, stream>>>(H, Hm);
  k_cqk<<<B_ * T_, 128, 0, stream>>>(Hm, Wq, Wk, CQ, CK);
  k_projq<<<(ROWS_ALL * NH_ + 255) / 256, 256, 0, stream>>>(Qbf, CQ, rho, projq, qrho);
  k_projk<<<(KL_ * ROWS_ALL * NH_ + 255) / 256, 256, 0, stream>>>(Kbf, CK, rho, projk);
  // ---- attention ----
  k_attn<<<KL_ * B_ * TR_, 256, 0, stream>>>(Qbf, Kbf, VbfT, projq, projk, qrho, logA,
                                             A, rsA, logrsA, beta, gamma_A, lambda_b,
                                             lag, msg, loss + 1);
  // ---- combine + output path ----
  k_combine<<<(ROWS_TR * D_ + 255) / 256, 256, 0, stream>>>(lag, msg, rsA, msgc, loss);
  k_gemm_mo<<<ROWS_TR / 16, 256, 0, stream>>>(msgc, WoT, mo);
  k_gate<<<(ROWS_TR + 255) / 256, 256, 0, stream>>>(H, mo, gate_w, gate_b, gm);
  k_ffn1<<<dim3(ROWS_TR / 16, 2), 256, 0, stream>>>(Hbf, gm, W1T, ffn_b1, ff1);
  k_ffn2_ln<<<ROWS_TR / 128, 256, 0, stream>>>(ff1, W2T, ffn_b2, H, ln_g, ln_b, out);
  k_head_loss<<<(B_ * KL_ * N_ * D_ + 255) / 256, 256, 0, stream>>>(H, loss, loss + 1, out);
}
